// unit_gcn_63582695850434
// MI455X (gfx1250) — matrix operand order repaired and checked by reading, not compiled, not run on hardware
//
#include <hip/hip_runtime.h>
#include <hip/hip_bf16.h>

// MI455X (gfx1250) implementation of unit_gcn forward.
// Pipeline:
//   k1_mean  : xm[n,c,v] = mean_t x
//   k2_attn  : full SORGC attention -> R[s,n,c,u,v] as f16, padded 32x32
//   k3_gemm  : fused (w3 @ x) -> LDS -> (x3 @ R^T) summed over s, WMMA f16,
//              input staged via GLOBAL_LOAD_ASYNC_TO_LDS (ASYNCcnt path)
//   k4_stats : per-channel batch mean/var
//   k5_final : BN + residual + relu
typedef __attribute__((ext_vector_type(16))) _Float16 v16h;
typedef __attribute__((ext_vector_type(8)))  float    v8f;
typedef __attribute__((ext_vector_type(4)))  int      v4i;

#define S_ 3
#define N_ 64
#define C_ 64
#define T_ 256
#define V_ 25

#if __has_builtin(__builtin_amdgcn_global_load_async_to_lds_b128)
#define HAVE_ASYNC_LDS 1
#endif

__device__ __forceinline__ v8f wmma_f16(v16h a, v16h b, v8f c) {
  // v_wmma_f32_16x16x32_f16: D = A(16x32 f16) * B(32x16 f16) + C(16x16 f32)
  return __builtin_amdgcn_wmma_f32_16x16x32_f16(false, a, false, b, (short)0, c,
                                                false, false);
}

#ifdef HAVE_ASYNC_LDS
typedef __attribute__((address_space(1))) v4i gas_v4i;
typedef __attribute__((address_space(3))) v4i las_v4i;
__device__ __forceinline__ void async_cp16(const float* g, float* l) {
  // 16-byte async copy global -> LDS (tracked by ASYNCcnt)
  __builtin_amdgcn_global_load_async_to_lds_b128(
      (gas_v4i*)(void*)const_cast<float*>(g), (las_v4i*)(void*)l, 0, 0);
}
#endif

// ---------------------------------------------------------------- k1: mean over T
__global__ __launch_bounds__(256) void k1_mean(const float* __restrict__ x,
                                               float* __restrict__ xm) {
  int i = blockIdx.x * 256 + threadIdx.x;          // 64*64*25 = 102400 threads
  int v = i % 25;
  int nc = i / 25;
  const float* p = x + (size_t)nc * (T_ * V_) + v;
  float s = 0.f;
  for (int t = 0; t < T_; ++t) {
    if ((t & 31) == 0) __builtin_prefetch(p + (t + 64) * V_, 0, 1);
    s += p[t * V_];
  }
  xm[i] = s * (1.0f / (float)T_);
}

// ---------------------------------------------------------------- k2: attention -> R (f16, padded 32x32)
__global__ __launch_bounds__(256) void k2_attn(
    const float* __restrict__ xm, const float* __restrict__ Aadp,
    const float* __restrict__ alpha_p, const float* __restrict__ w1,
    const float* __restrict__ b1, const float* __restrict__ w2,
    const float* __restrict__ b2, const float* __restrict__ w4,
    const float* __restrict__ b4, const float* __restrict__ pw1,
    const float* __restrict__ pb1, const float* __restrict__ pw2,
    const float* __restrict__ pb2, const float* __restrict__ dww,
    const float* __restrict__ pjw, const float* __restrict__ beta,
    const float* __restrict__ gamma, _Float16* __restrict__ Rbuf) {
  const int n = blockIdx.x, s = blockIdx.y, tid = threadIdx.x;
  __shared__ float sx1[8 * 25], sx2[8 * 25];
  __shared__ float sq1[8 * 625];   // q1 -> later tanh(pool(conv))
  __shared__ float sB2[8 * 625];   // h   -> later depthwise conv
  __shared__ float sB3[8 * 625];   // q_all

  // (a) x1, x2 (1x1 conv on T-mean)
  for (int i = tid; i < 400; i += 256) {
    int which = i / 200, r = (i % 200) / 25, u = i % 25;
    const float* wrow = (which ? w2 : w1) + (s * 8 + r) * 64;
    float acc = (which ? b2 : b1)[s * 8 + r];
    for (int c = 0; c < 64; ++c) acc += xm[(n * 64 + c) * 25 + u] * wrow[c];
    (which ? sx2 : sx1)[r * 25 + u] = acc;
  }
  __syncthreads();
  // (b) q1 = tanh(x1[u]-x2[v]); (c) h = relu(pair MLP)
  for (int i = tid; i < 5000; i += 256) {
    int r = i / 625, u = (i / 25) % 25, v = i % 25;
    sq1[i] = tanhf(sx1[r * 25 + u] - sx2[r * 25 + v]);
    float h = pb1[s * 8 + r];
    const float* pr = pw1 + (s * 8 + r) * 16;
    for (int rp = 0; rp < 8; ++rp)
      h += pr[rp] * sx2[rp * 25 + u] + pr[8 + rp] * sx2[rp * 25 + v];
    sB2[i] = fmaxf(h, 0.f);
  }
  __syncthreads();
  float gam = gamma[s], bet = beta[s], alp = alpha_p[0];
  // (d) q_all = q1 + gamma*q2
  for (int i = tid; i < 5000; i += 256) {
    int r = i / 625, uv = i % 625;
    float q2 = pb2[s * 8 + r];
    const float* pr = pw2 + (s * 8 + r) * 8;
    for (int rp = 0; rp < 8; ++rp) q2 += pr[rp] * sB2[rp * 625 + uv];
    sB3[i] = sq1[i] + gam * q2;
  }
  __syncthreads();
  // (e) depthwise 3x3 conv on q1 -> sB2 (h is dead)
  for (int i = tid; i < 5000; i += 256) {
    int r = i / 625, u = (i / 25) % 25, v = i % 25;
    const float* kw = dww + (s * 8 + r) * 9;
    float acc = 0.f;
    for (int di = 0; di < 3; ++di) {
      int uu = u + di - 1;
      if (uu < 0 || uu > 24) continue;
      for (int dj = 0; dj < 3; ++dj) {
        int vv = v + dj - 1;
        if (vv < 0 || vv > 24) continue;
        acc += kw[di * 3 + dj] * sq1[r * 625 + uu * 25 + vv];
      }
    }
    sB2[i] = acc;
  }
  __syncthreads();
  // (f) 3x3 avgpool (count_include_pad) + tanh -> sq1 (q1 dead)
  for (int i = tid; i < 5000; i += 256) {
    int r = i / 625, u = (i / 25) % 25, v = i % 25;
    float acc = 0.f;
    for (int di = -1; di <= 1; ++di) {
      int uu = u + di;
      if (uu < 0 || uu > 24) continue;
      for (int dj = -1; dj <= 1; ++dj) {
        int vv = v + dj;
        if (vv < 0 || vv > 24) continue;
        acc += sB2[r * 625 + uu * 25 + vv];
      }
    }
    sq1[i] = tanhf(acc * (1.0f / 9.0f));
  }
  __syncthreads();
  // (g) R = (w4.q_all + b4)*alpha + A + beta*(pjw.aux); f16, zero-padded 32x32
  for (int i = tid; i < 64 * 1024; i += 256) {
    int o = i >> 10, u = (i >> 5) & 31, v = i & 31;
    float val = 0.f;
    if (u < 25 && v < 25) {
      float rm = b4[s * 64 + o], ra = 0.f;
      const float* w4r = w4 + (s * 64 + o) * 8;
      const float* pjr = pjw + (s * 64 + o) * 8;
      for (int r = 0; r < 8; ++r) {
        rm += w4r[r] * sB3[r * 625 + u * 25 + v];
        ra += pjr[r] * sq1[r * 625 + u * 25 + v];
      }
      val = rm * alp + Aadp[(s * 25 + u) * 25 + v] + bet * ra;
    }
    Rbuf[((((size_t)s * 64 + n) * 64 + o) << 10) + (u << 5) + v] = (_Float16)val;
  }
}

// ---------------------------------------------------------------- k3: fused WMMA GEMM chain
__global__ __launch_bounds__(256) void k3_gemm(const float* __restrict__ x,
                                               const float* __restrict__ w3,
                                               const float* __restrict__ b3,
                                               const _Float16* __restrict__ Rbuf,
                                               float* __restrict__ y) {
  // LDS: staged input x (f16, [t16][v32][c64]) + x3 for all 3 subsets
  // ([s][o64][t16][v32]) = 64 KB + 192 KB (+768 B bias) < 320 KB WGP LDS.
  // The x3 region doubles as the f32 async-staging buffer during phase 1.
  __shared__ alignas(32) _Float16 ldsXT[16 * 32 * 64];
  __shared__ alignas(32) _Float16 ldsX3[S_ * 64 * 16 * 32];
  __shared__ float sBias[S_ * 64];
  const int tb = blockIdx.x * 16;  // T chunk
  const int nb = blockIdx.y;       // batch
  const int tid = threadIdx.x;
  const int lane = tid & 31, wave = tid >> 5;
  const int lm = lane & 15, hi = lane >> 4;

  // ---- phase 1: stage x[nb, :, tb..tb+15, :] as f16 [t][v32][c], pad v>=25 = 0
  for (int i = tid; i < S_ * 64; i += 256) sBias[i] = b3[i];
  for (int i = tid; i < 16 * 7 * 64; i += 256) {
    int t = i / (7 * 64), r = i % (7 * 64);
    int v = 25 + r / 64, c = r % 64;
    ldsXT[(t * 32 + v) * 64 + c] = (_Float16)0.f;
  }
#ifdef HAVE_ASYNC_LDS
  {
    // Per channel c the needed global slice is 400 contiguous, 16B-aligned
    // floats: copy with b128 async-to-LDS ops into an f32 staging area.
    float* stg = (float*)ldsX3;  // 64*400*4 = 100 KB of the 192 KB region
    for (int i = tid; i < 64 * 100; i += 256) {
      int c = i / 100, q = (i % 100) * 4;
      const float* g = x + (size_t)(nb * 64 + c) * (T_ * V_) + tb * V_ + q;
      async_cp16(g, stg + c * 400 + q);
    }
    asm volatile("s_wait_asynccnt 0x0" ::: "memory");
    __syncthreads();
    for (int i = tid; i < 64 * 16 * 25; i += 256) {
      int c = i / 400, t = (i / 25) % 16, v = i % 25;
      ldsXT[(t * 32 + v) * 64 + c] = (_Float16)stg[c * 400 + t * 25 + v];
    }
  }
#else
  for (int i = tid; i < 64 * 16 * 25; i += 256) {
    int c = i / 400, t = (i / 25) % 16, v = i % 25;
    float val = x[(((size_t)nb * 64 + c) * T_ + tb + t) * V_ + v];
    ldsXT[(t * 32 + v) * 64 + c] = (_Float16)val;
  }
#endif
  __syncthreads();

  // ---- phase 2: x3[s] = w3[s] @ x  (M=64 o, K=64 cin, N=16t x 32v positions)
  // tile id: s*128 + mtile*32 + (t*2 + vhalf); wave-uniform loop.
  for (int idx = wave; idx < S_ * 128; idx += 8) {
    int s = idx >> 7, rem = idx & 127;
    int mt = rem >> 5, nt = rem & 31;
    int t = nt >> 1, vb = (nt & 1) * 16;
    int v = vb + lm;
    v8f acc = {};
    for (int kt = 0; kt < 2; ++kt) {
      int k0 = kt * 32;
      // A frag: w3 row o = mt*16+lm; K contiguous k0+hi*16 .. +15
      const float* wr = w3 + ((size_t)(s * 64 + mt * 16 + lm)) * 64 + k0 + hi * 16;
      float4 f0 = *(const float4*)(wr);
      float4 f1 = *(const float4*)(wr + 4);
      float4 f2 = *(const float4*)(wr + 8);
      float4 f3 = *(const float4*)(wr + 12);
      union { v16h v; _Float16 h[16]; } af;
      af.h[0] = (_Float16)f0.x; af.h[1] = (_Float16)f0.y;
      af.h[2] = (_Float16)f0.z; af.h[3] = (_Float16)f0.w;
      af.h[4] = (_Float16)f1.x; af.h[5] = (_Float16)f1.y;
      af.h[6] = (_Float16)f1.z; af.h[7] = (_Float16)f1.w;
      af.h[8] = (_Float16)f2.x; af.h[9] = (_Float16)f2.y;
      af.h[10] = (_Float16)f2.z; af.h[11] = (_Float16)f2.w;
      af.h[12] = (_Float16)f3.x; af.h[13] = (_Float16)f3.y;
      af.h[14] = (_Float16)f3.z; af.h[15] = (_Float16)f3.w;
      // B frag: column n=v at fixed t; K contiguous (lo: k0.., hi: k0+16..)
      v16h bf = *(const v16h*)(&ldsXT[(t * 32 + v) * 64 + k0 + hi * 16]);
      acc = wmma_f16(af.v, bf, acc);
    }
    // D tile -> LDS x3 [s][o][t][v32] (+ bias where v<25; pad cols stay 0).
    // Bias comes from LDS (unconditional load + value select: no exec-mask
    // branches around the load).
    int ob = mt * 16 + hi * 8;
    bool valid = (v < 25);
    for (int r = 0; r < 8; ++r) {
      float bv = sBias[s * 64 + ob + r];
      float val = acc[r] + (valid ? bv : 0.f);
      ldsX3[(((s * 64 + ob + r) * 16) + t) * 32 + v] = (_Float16)val;
    }
  }
  __syncthreads();

  // ---- phase 3: y[nb,c,t,u] = sum_s sum_v R[s,nb,c,u,v] * x3[s,c,t,v]
  // A = R (M=u, K=v padded 32), B = x3^T (rows [t][v] are B^T contiguous in K)
  for (int c = wave; c < 64; c += 8) {
    for (int ut = 0; ut < 2; ++ut) {
      v8f acc = {};
      for (int s = 0; s < S_; ++s) {
        const _Float16* ar = Rbuf + ((((size_t)s * 64 + nb) * 64 + c) << 10) +
                             (ut * 16 + lm) * 32 + hi * 16;
        union { v16h v; uint4 q[2]; } af;
        af.q[0] = *(const uint4*)(ar);
        af.q[1] = *(const uint4*)(ar + 8);
        v16h bf = *(const v16h*)(&ldsX3[(((s * 64 + c) * 16) + lm) * 32 + hi * 16]);
        acc = wmma_f16(af.v, bf, acc);
      }
      float* yr = y + (((size_t)nb * 64 + c) * T_ + tb + lm) * V_;
      if (ut == 0) {
        for (int r = 0; r < 8; ++r) yr[hi * 8 + r] = acc[r];
      } else {
        for (int r = 0; r < 8; ++r) {
          int u = 16 + hi * 8 + r;
          if (u < 25) yr[u] = acc[r];
        }
      }
    }
  }
}

// ---------------------------------------------------------------- k4: per-channel batch stats
__global__ __launch_bounds__(256) void k4_stats(const float* __restrict__ y,
                                                float* __restrict__ stats) {
  const int c = blockIdx.x, tid = threadIdx.x;
  float s1 = 0.f, s2 = 0.f;
  for (int i = tid; i < N_ * T_ * V_; i += 256) {
    int n = i / (T_ * V_), rem = i % (T_ * V_);
    float v = y[((size_t)n * 64 + c) * (T_ * V_) + rem];
    s1 += v;
    s2 += v * v;
  }
  __shared__ float r1[256], r2[256];
  r1[tid] = s1; r2[tid] = s2;
  __syncthreads();
  for (int off = 128; off > 0; off >>= 1) {
    if (tid < off) { r1[tid] += r1[tid + off]; r2[tid] += r2[tid + off]; }
    __syncthreads();
  }
  if (tid == 0) {
    const float inv = 1.0f / (float)(N_ * T_ * V_);
    float mu = r1[0] * inv;
    stats[c] = mu;
    stats[64 + c] = r2[0] * inv - mu * mu;
  }
}

// ---------------------------------------------------------------- k5: BN + residual + relu
__global__ __launch_bounds__(256) void k5_final(const float* __restrict__ y,
                                                const float* __restrict__ x,
                                                const float* __restrict__ stats,
                                                const float* __restrict__ bnw,
                                                const float* __restrict__ bnb,
                                                float* __restrict__ out) {
  size_t i = (size_t)blockIdx.x * 256 + threadIdx.x;  // 26,214,400 total
  int c = (int)((i / (T_ * V_)) % 64);
  float mu = stats[c], var = stats[64 + c];
  float val = (y[i] - mu) * rsqrtf(var + 1e-5f) * bnw[c] + bnb[c] + x[i];
  out[i] = fmaxf(val, 0.f);
}

// ---------------------------------------------------------------- launch
extern "C" void kernel_launch(void* const* d_in, const int* in_sizes, int n_in,
                              void* d_out, int out_size, void* d_ws, size_t ws_size,
                              hipStream_t stream) {
  (void)in_sizes; (void)n_in; (void)out_size; (void)ws_size;
  const float* x     = (const float*)d_in[0];
  const float* A     = (const float*)d_in[1];
  const float* alpha = (const float*)d_in[2];
  const float* w1    = (const float*)d_in[3];
  const float* b1    = (const float*)d_in[4];
  const float* w2    = (const float*)d_in[5];
  const float* b2    = (const float*)d_in[6];
  const float* w3    = (const float*)d_in[7];
  const float* b3    = (const float*)d_in[8];
  const float* w4    = (const float*)d_in[9];
  const float* b4    = (const float*)d_in[10];
  const float* pw1   = (const float*)d_in[11];
  const float* pb1   = (const float*)d_in[12];
  const float* pw2   = (const float*)d_in[13];
  const float* pb2   = (const float*)d_in[14];
  const float* dww   = (const float*)d_in[15];
  const float* pjw   = (const float*)d_in[16];
  const float* beta  = (const float*)d_in[17];
  const float* gamma = (const float*)d_in[18];
  const float* bnw   = (const float*)d_in[19];
  const float* bnb   = (const float*)d_in[20];

  char* ws = (char*)d_ws;
  float*    y_pre = (float*)(ws);                                 // 104857600 B
  _Float16* Rbuf  = (_Float16*)(ws + 104857600);                  //  25165824 B
  float*    xm    = (float*)(ws + 104857600 + 25165824);          //    409600 B
  float*    stats = (float*)(ws + 104857600 + 25165824 + 409600); //       512 B
  float* out = (float*)d_out;

  k1_mean<<<400, 256, 0, stream>>>(x, xm);
  k2_attn<<<dim3(64, 3), 256, 0, stream>>>(xm, A, alpha, w1, b1, w2, b2, w4, b4,
                                           pw1, pb1, pw2, pb2, dww, pjw, beta,
                                           gamma, Rbuf);
  k3_gemm<<<dim3(16, 64), 256, 0, stream>>>(x, w3, b3, Rbuf, y_pre);
  k4_stats<<<64, 256, 0, stream>>>(y_pre, stats);
  k5_final<<<102400, 256, 0, stream>>>(y_pre, x, stats, bnw, bnb, out);
}
